// UncertaintyRouter_75024488727159
// MI455X (gfx1250) — compile-verified, matrix operand order checked by reading, not run on hardware
//
#include <hip/hip_runtime.h>

// Problem constants (from reference): N=8192, D=4096, E=64, K=8, H=1024
#define N_TOK 8192
#define DIMZ  4096
#define HID   1024
#define NE    64
#define TOPK  8

typedef __attribute__((ext_vector_type(16))) __bf16 v16bf;
typedef __attribute__((ext_vector_type(8)))  float  v8f;
typedef __attribute__((ext_vector_type(4)))  float  vf4;
typedef __attribute__((ext_vector_type(4)))  __bf16 v4bf;
typedef __attribute__((ext_vector_type(4)))  unsigned int u32x4;
typedef __attribute__((ext_vector_type(4)))  int i32x4;
typedef __attribute__((ext_vector_type(8)))  int i32x8;

#if defined(__has_builtin)
#if __has_builtin(__builtin_amdgcn_tensor_load_to_lds) && \
    __has_builtin(__builtin_amdgcn_s_wait_tensorcnt)
#define HAVE_TDM 1
#endif
#if __has_builtin(__builtin_amdgcn_sched_group_barrier)
#define SGB(m, n, id) __builtin_amdgcn_sched_group_barrier((m), (n), (id))
#endif
#endif
#ifndef SGB
#define SGB(m, n, id)
#endif

union FragBF {
    v16bf v;
    uint4 q[2];
};

// ---------------------------------------------------------------------------
// 1) fp32 -> bf16 bulk convert (for z)
// ---------------------------------------------------------------------------
__global__ void cvt_f32_bf16(const float* __restrict__ in,
                             __bf16* __restrict__ out, long n) {
    long i = ((long)blockIdx.x * blockDim.x + threadIdx.x) * 4;
    long stride = (long)gridDim.x * blockDim.x * 4;
    for (long o = i; o < n; o += stride) {
        vf4 f = *(const vf4*)(in + o);
        v4bf b = __builtin_convertvector(f, v4bf);
        *(v4bf*)(out + o) = b;
    }
}

// ---------------------------------------------------------------------------
// 2) fp32 [R x C] -> bf16 transposed [C x R]
// ---------------------------------------------------------------------------
__global__ void transp_cvt(const float* __restrict__ in,
                           __bf16* __restrict__ out, int R, int C) {
    long total = (long)R * C;
    long stride = (long)gridDim.x * blockDim.x;
    for (long o = (long)blockIdx.x * blockDim.x + threadIdx.x; o < total; o += stride) {
        long c = o / R;
        long r = o - c * R;
        out[o] = (__bf16)in[r * (long)C + c];
    }
}

// ---------------------------------------------------------------------------
// TDM: issue a 2D tile DMA (global -> LDS) via the Tensor Data Mover.
// D# layout per CDNA5 ISA ch.8: group0 = {count, lds_addr, global_addr, type=2},
// group1 = {wg_mask=0, data_size, tensor_dim0/1, tile_dim0/1, dim0_stride}.
// data_size=1 (2-byte elements). Groups 2/3 unused (2D tensor).
// ---------------------------------------------------------------------------
#if HAVE_TDM
__device__ __forceinline__ void tdm_load_tile_2d(unsigned ldsOff,
                                                 const void* gtile,
                                                 unsigned tile0, unsigned tile1,
                                                 unsigned long long ten0,
                                                 unsigned long long ten1,
                                                 unsigned long long stride0) {
    union { u32x4 v; unsigned u[4]; } g0;
    union { i32x8 v; unsigned long long q[4]; } g1;
    unsigned long long ga = (unsigned long long)(uintptr_t)gtile;
    g0.u[0] = 1u;                                         // count=1 (valid user D#)
    g0.u[1] = ldsOff;                                     // lds_addr (bytes)
    g0.u[2] = (unsigned)(ga & 0xffffffffull);             // global_addr[31:0]
    g0.u[3] = (unsigned)((ga >> 32) & 0x01ffffffull)      // global_addr[56:32]
            | 0x80000000u;                                // type=2 ("image")
    g1.q[0] = (1ull << 16)                                // data_size=1 -> 2 bytes
            | ((ten0 & 0xffffull) << 48);                 // tensor_dim0[15:0]
    g1.q[1] = ((ten0 >> 16) & 0xffffull)                  // tensor_dim0[31:16]
            | ((ten1 & 0xffffull) << 16)                  // tensor_dim1[15:0]
            | (((ten1 >> 16) & 0xffffull) << 32)          // tensor_dim1[31:16]
            | ((unsigned long long)(tile0 & 0xffffu) << 48); // tile_dim0
    g1.q[2] = (unsigned long long)(tile1 & 0xffffu)       // tile_dim1
            | ((stride0 & 0xffffffffull) << 32);          // dim0_stride[31:0]
    g1.q[3] = (stride0 >> 32) & 0xffffull;                // dim0_stride[47:32]
    i32x4 gz = {0, 0, 0, 0};
#if __clang_major__ >= 23
    i32x8 gz8 = {0, 0, 0, 0, 0, 0, 0, 0};
    __builtin_amdgcn_tensor_load_to_lds(g0.v, g1.v, gz, gz, gz8, 0);
#else
    __builtin_amdgcn_tensor_load_to_lds(g0.v, g1.v, gz, gz, 0);
#endif
}
#endif

// ---------------------------------------------------------------------------
// Compute one 16(M)x128(cols)x64(K) slab.
// A fragments fetched up-front in one vmem clause. B fragments run through a
// 3-buffer rotation (2 fragments in flight), and sched_group_barrier pins the
// shape: [4 vmem][4 ds] then [1 wmma][2 ds] x14 then [2 wmma], so each wmma
// waits dscnt<=4 instead of draining the DS queue to zero.
// ---------------------------------------------------------------------------
__device__ __forceinline__ void mma_slab(const __bf16* __restrict__ zb,
                                         const __bf16* Btile,
                                         int m0, int laneM, int khalf, int kk,
                                         v8f acc[8]) {
    FragBF a[2];
    const uint4* pa =
        (const uint4*)(zb + (size_t)(m0 + laneM) * DIMZ + kk + khalf);
    a[0].q[0] = pa[0];
    a[0].q[1] = pa[2];
    a[1].q[0] = pa[4];
    a[1].q[1] = pa[6];
    __builtin_prefetch((const void*)(pa + 8), 0, 1);   // next K slab of z

    #pragma unroll
    for (int kb = 0; kb < 2; ++kb) {
        FragBF b[3];
        #pragma unroll
        for (int p = 0; p < 2; ++p) {
            const uint4* pb = (const uint4*)(
                Btile + (p * 16 + laneM) * 64 + kb * 32 + khalf);
            b[p].q[0] = pb[0];
            b[p].q[1] = pb[2];
        }
        #pragma unroll
        for (int c = 0; c < 8; ++c) {
            if (c + 2 < 8) {
                const uint4* pb = (const uint4*)(
                    Btile + ((c + 2) * 16 + laneM) * 64 + kb * 32 + khalf);
                b[(c + 2) % 3].q[0] = pb[0];
                b[(c + 2) % 3].q[1] = pb[2];
            }
            acc[c] = __builtin_amdgcn_wmma_f32_16x16x32_bf16(
                false, a[kb].v, false, b[c % 3].v, (short)0, acc[c],
                false, false);
        }
    }

    // Scheduling pins for this region: totals = 4 vmem-read, 32 ds-read,
    // 16 wmma.  Keep the DS pipeline 2 fragments ahead of the wmma stream.
    SGB(0x020, 4, 0);                 // 4x global_load_b128 (A fragments)
    SGB(0x100, 4, 0);                 // first two B fragments
    #pragma unroll
    for (int i = 0; i < 14; ++i) {
        SGB(0x008, 1, 0);             // one wmma
        SGB(0x100, 2, 0);             // next B fragment (2 ds reads)
    }
    SGB(0x008, 2, 0);                 // last two wmma
}

// ---------------------------------------------------------------------------
// 3) GEMM1: h = relu(z @ W1 + b1), both branches (blockIdx.z).
//    Block tile 128(M) x 128(Hcols), 8 waves, wave owns 16 rows x 128 cols.
//    Weight tile 128x64 double-buffered in LDS, filled by the Tensor Data
//    Mover (tensor_load_to_lds + s_wait_tensorcnt), overlapped with WMMA.
// ---------------------------------------------------------------------------
__global__ void gemm1_wmma(const __bf16* __restrict__ zb,
                           const __bf16* __restrict__ w1ts,
                           const __bf16* __restrict__ w1tn,
                           const float* __restrict__ bs1,
                           const float* __restrict__ bn1,
                           __bf16* __restrict__ hs,
                           __bf16* __restrict__ hn) {
    __shared__ __align__(16) __bf16 Bt[2][128 * 64];   // 2 x 16 KB

    const int branch = blockIdx.z;
    const __bf16* wt = branch ? w1tn : w1ts;
    const float* bias = branch ? bn1 : bs1;
    __bf16* hout = branch ? hn : hs;

    const int colBase = blockIdx.x * 128;
    const int rowBase = blockIdx.y * 128;
    const int tid  = threadIdx.x;
    const int wv   = tid >> 5;
    const int lane = tid & 31;
    const int laneM = lane & 15;
    const int khalf = (lane >> 4) << 3;   // 0 or 8
    const int m0 = rowBase + wv * 16;

    v8f acc[8] = {};

#if HAVE_TDM
    const __bf16* wtile = wt + (size_t)colBase * DIMZ;
    if (tid == 0) {
        tdm_load_tile_2d((unsigned)(uintptr_t)&Bt[0][0], wtile,
                         64, 128, DIMZ, HID, DIMZ);
        __builtin_amdgcn_s_wait_tensorcnt(0);
    }
    __syncthreads();

    int cur = 0;
    for (int kk = 0; kk < DIMZ; kk += 64) {
        if (tid == 0 && kk + 64 < DIMZ) {
            // prefetch next weight tile into the other LDS buffer (async DMA)
            tdm_load_tile_2d((unsigned)(uintptr_t)&Bt[cur ^ 1][0],
                             wtile + kk + 64, 64, 128, DIMZ, HID, DIMZ);
        }
        mma_slab(zb, &Bt[cur][0], m0, laneM, khalf, kk, acc);
        if (tid == 0) __builtin_amdgcn_s_wait_tensorcnt(0);
        __syncthreads();
        cur ^= 1;
    }
#else
    for (int kk = 0; kk < DIMZ; kk += 64) {
        __syncthreads();
        // cooperative stage: 128 rows x 64 bf16 = 1024 uint4, 256 threads
        #pragma unroll
        for (int it = 0; it < 4; ++it) {
            int q = tid + it * 256;
            int row = q >> 3;
            int part = q & 7;
            const uint4* src =
                (const uint4*)(wt + (size_t)(colBase + row) * DIMZ + kk + part * 8);
            ((uint4*)&Bt[0][0])[q] = *src;
            __builtin_prefetch((const void*)(src + 16), 0, 1);
        }
        __syncthreads();
        mma_slab(zb, &Bt[0][0], m0, laneM, khalf, kk, acc);
    }
#endif

    // epilogue: +bias, relu, store bf16
    #pragma unroll
    for (int c = 0; c < 8; ++c) {
        const int cc = colBase + c * 16 + laneM;
        const float bval = bias[cc];
        #pragma unroll
        for (int r = 0; r < 8; ++r) {
            const int rr = m0 + r + ((lane >> 4) << 3);
            float v = acc[c][r] + bval;
            v = v > 0.f ? v : 0.f;
            hout[(size_t)rr * HID + cc] = (__bf16)v;
        }
    }
}

// ---------------------------------------------------------------------------
// 4) GEMM2 + gating. One block per 16 rows. 128 threads = 4 waves.
// ---------------------------------------------------------------------------
__global__ void gemm2_gate(const __bf16* __restrict__ hs,
                           const __bf16* __restrict__ hn,
                           const __bf16* __restrict__ w2ts,
                           const __bf16* __restrict__ w2tn,
                           const float* __restrict__ bs2,
                           const float* __restrict__ bn2,
                           const float* __restrict__ eps,
                           float* __restrict__ out) {
    __shared__ float sS[16][NE];
    __shared__ float sN[16][NE];

    const int row0 = blockIdx.x * 16;
    const int tid  = threadIdx.x;
    const int wv   = tid >> 5;
    const int lane = tid & 31;
    const int laneM = lane & 15;
    const int khalf = (lane >> 4) << 3;

    const int branch = wv >> 1;
    const int ctb = (wv & 1) * 2;
    const __bf16* hb = branch ? hn : hs;
    const __bf16* w2 = branch ? w2tn : w2ts;
    const float* bias = branch ? bn2 : bs2;

    v8f acc0 = {}, acc1 = {};

    #pragma unroll
    for (int kk = 0; kk < HID; kk += 32) {
        FragBF a, b0, b1;
        const uint4* pa =
            (const uint4*)(hb + (size_t)(row0 + laneM) * HID + kk + khalf);
        const uint4* pb0 =
            (const uint4*)(w2 + (size_t)(ctb * 16 + laneM) * HID + kk + khalf);
        const uint4* pb1 =
            (const uint4*)(w2 + (size_t)((ctb + 1) * 16 + laneM) * HID + kk + khalf);
        a.q[0]  = pa[0];  a.q[1]  = pa[2];
        b0.q[0] = pb0[0]; b0.q[1] = pb0[2];
        b1.q[0] = pb1[0]; b1.q[1] = pb1[2];
        __builtin_prefetch((const void*)(pa + 4), 0, 1);

        acc0 = __builtin_amdgcn_wmma_f32_16x16x32_bf16(
            false, a.v, false, b0.v, (short)0, acc0, false, false);
        acc1 = __builtin_amdgcn_wmma_f32_16x16x32_bf16(
            false, a.v, false, b1.v, (short)0, acc1, false, false);
    }

    float* dst = branch ? &sN[0][0] : &sS[0][0];
    {
        const int cc0 = ctb * 16 + laneM;
        const int cc1 = (ctb + 1) * 16 + laneM;
        const float bv0 = bias[cc0];
        const float bv1 = bias[cc1];
        #pragma unroll
        for (int r = 0; r < 8; ++r) {
            const int rr = r + ((lane >> 4) << 3);
            dst[rr * NE + cc0] = acc0[r] + bv0;
            dst[rr * NE + cc1] = acc1[r] + bv1;
        }
    }
    __syncthreads();

    // ---- gating: each wave handles 4 rows; 64 experts = 2 per lane ----
    const float NEG = -3.0e38f;
    for (int i = 0; i < 4; ++i) {
        const int rr = wv * 4 + i;
        const int rowg = row0 + rr;
        const int e0 = lane;
        const int e1 = lane + 32;

        float n0 = sN[rr][e0], n1 = sN[rr][e1];
        float sp0 = (n0 > 20.f) ? n0 : log1pf(__expf(n0));
        float sp1 = (n1 > 20.f) ? n1 : log1pf(__expf(n1));
        float f0 = sS[rr][e0] + sp0 * eps[(size_t)rowg * NE + e0];
        float f1 = sS[rr][e1] + sp1 * eps[(size_t)rowg * NE + e1];

        float tv[TOPK];
        int   ti[TOPK];
        float l0 = f0, l1 = f1;
        #pragma unroll
        for (int t = 0; t < TOPK; ++t) {
            float bv; int bi;
            if (l0 >= l1) { bv = l0; bi = e0; } else { bv = l1; bi = e1; }
            #pragma unroll
            for (int off = 16; off > 0; off >>= 1) {
                float ov = __shfl_xor(bv, off, 32);
                int   oi = __shfl_xor(bi, off, 32);
                if (ov > bv || (ov == bv && oi < bi)) { bv = ov; bi = oi; }
            }
            tv[t] = bv; ti[t] = bi;
            if (bi == e0) l0 = NEG;
            if (bi == e1) l1 = NEG;
        }

        const float m = tv[0];
        float sum = 0.f;
        #pragma unroll
        for (int t = 0; t < TOPK; ++t) sum += __expf(tv[t] - m);
        const float inv = 1.f / sum;

        float w0 = 0.f, w1 = 0.f;
        #pragma unroll
        for (int t = 0; t < TOPK; ++t) {
            const float w = __expf(tv[t] - m) * inv;
            if (ti[t] == e0) w0 = w;
            if (ti[t] == e1) w1 = w;
        }
        out[(size_t)rowg * NE + e0] = w0;
        out[(size_t)rowg * NE + e1] = w1;

        if (lane == 0) {
            float* oidx = out + (size_t)N_TOK * NE + (size_t)rowg * TOPK;
            #pragma unroll
            for (int t = 0; t < TOPK; ++t) oidx[t] = (float)ti[t];
        }
    }
}

// ---------------------------------------------------------------------------
extern "C" void kernel_launch(void* const* d_in, const int* in_sizes, int n_in,
                              void* d_out, int out_size, void* d_ws, size_t ws_size,
                              hipStream_t stream) {
    (void)in_sizes; (void)n_in; (void)out_size; (void)ws_size;

    const float* z   = (const float*)d_in[0];
    const float* eps = (const float*)d_in[1];
    const float* ws1 = (const float*)d_in[2];
    const float* bs1 = (const float*)d_in[3];
    const float* ws2 = (const float*)d_in[4];
    const float* bs2 = (const float*)d_in[5];
    const float* wn1 = (const float*)d_in[6];
    const float* bn1 = (const float*)d_in[7];
    const float* wn2 = (const float*)d_in[8];
    const float* bn2 = (const float*)d_in[9];
    float* out = (float*)d_out;

    // workspace layout (bf16 elements), ~118 MB total
    __bf16* zb   = (__bf16*)d_ws;
    __bf16* w1ts = zb   + (size_t)N_TOK * DIMZ;
    __bf16* w1tn = w1ts + (size_t)HID * DIMZ;
    __bf16* w2ts = w1tn + (size_t)HID * DIMZ;
    __bf16* w2tn = w2ts + (size_t)NE * HID;
    __bf16* hs   = w2tn + (size_t)NE * HID;
    __bf16* hn   = hs   + (size_t)N_TOK * HID;

    cvt_f32_bf16<<<2048, 256, 0, stream>>>(z, zb, (long)N_TOK * DIMZ);
    transp_cvt<<<1024, 256, 0, stream>>>(ws1, w1ts, DIMZ, HID);
    transp_cvt<<<1024, 256, 0, stream>>>(wn1, w1tn, DIMZ, HID);
    transp_cvt<<<64,   256, 0, stream>>>(ws2, w2ts, HID, NE);
    transp_cvt<<<64,   256, 0, stream>>>(wn2, w2tn, HID, NE);

    dim3 g1(HID / 128, N_TOK / 128, 2);
    gemm1_wmma<<<g1, 256, 0, stream>>>(zb, w1ts, w1tn, bs1, bn1, hs, hn);

    gemm2_gate<<<N_TOK / 16, 128, 0, stream>>>(hs, hn, w2ts, w2tn,
                                               bs2, bn2, eps, out);
}
